// DWTModelAllBands_39943195853366
// MI455X (gfx1250) — matrix operand order repaired and checked by reading, not compiled, not run on hardware
//
#include <hip/hip_runtime.h>
#include <stdint.h>

// ---------------------------------------------------------------------------
// DWT roundtrip, algebraically simplified: the level-2 Haar analysis+synthesis
// cancels exactly (orthogonal), so
//   out = ihaar2( LL1, B(lh1), B(hl1), B(hh1) ),  B = cubic_up2 ∘ area_down2
// (separable banded blur, edge-replicated on the 128-res grid).
// One workgroup (8 wave32) per 64x64 output tile:
//   stage 1: global_load_async_to_lds_b128 stages 80x80 input region into LDS
//   stage 2: level-1 Haar + 2x2 mean -> f16 down-band tiles (WMMA-B layout)
//   stage 3/4: separable resampling as banded-matrix WMMAs (f16 in, f32 acc),
//              constant weight matrices pre-baked in __constant__ memory
//   stage 5: inverse Haar, coalesced float2 stores
// ---------------------------------------------------------------------------

typedef __attribute__((ext_vector_type(16))) _Float16 v16h;
typedef __attribute__((ext_vector_type(8)))  _Float16 v8h;
typedef __attribute__((ext_vector_type(8)))  float    v8f;

#define SQ   0.70710678118654752f   // 1/sqrt(2)
#define IMGW 512
#define DH   20                      // down-band tile incl. halo of 2
#define INR  80                      // staged input rows/cols
#define INS  84                      // staged input stride (floats, 16B mult)
#define HSTR 40                      // f16 tile row stride (80B: 16B-mult, bank-spread)
#define TS   33                      // 32x32 f32 tile stride

// cv2 INTER_CUBIC (a=-0.75), half-pixel-center, exact 2x taps
constexpr float cubw_c(int t) {
  return t == 0 ? -0.03515625f : t == 1 ? 0.26171875f
       : t == 2 ?  0.87890625f : -0.10546875f;
}
// weight of local down-grid index p (halo of 2 included) for output i in [0,32)
constexpr float upw_c(int i, int p) {
  int k = i >> 1;
  if ((i & 1) == 0) { int t = p - k;     return (t >= 0 && t < 4) ? cubw_c(t)     : 0.f; }
  else              { int t = p - k - 1; return (t >= 0 && t < 4) ? cubw_c(3 - t) : 0.f; }
}

struct __align__(32) HTab { _Float16 v[2][32][16]; };

// A-operand table for pass 1 (rows): 16x32 f16 A layout, contiguous k = 16*(lane/16) + e
constexpr HTab makeWrA() {
  HTab h{};
  for (int cr = 0; cr < 2; ++cr)
    for (int lane = 0; lane < 32; ++lane)
      for (int e = 0; e < 16; ++e) {
        int i  = (lane & 15) + 16 * cr;
        int kb = (lane < 16) ? 0 : 16;
        int K  = kb + e;
        h.v[cr][lane][e] = (_Float16)upw_c(i, K);
      }
  return h;
}
// B-operand table for pass 2 (cols): 32x16 f16 B layout
constexpr HTab makeWcB() {
  HTab h{};
  for (int cc = 0; cc < 2; ++cc)
    for (int lane = 0; lane < 32; ++lane)
      for (int e = 0; e < 16; ++e) {
        int j  = (lane & 15) + 16 * cc;
        int Kb = (lane < 16) ? 0 : 8;
        h.v[cc][lane][e] = (_Float16)upw_c(j, Kb + e + ((e >= 8) ? 8 : 0));
      }
  return h;
}
__constant__ HTab cWrA = makeWrA();
__constant__ HTab cWcB = makeWcB();

__global__ __launch_bounds__(256)
void dwt_roundtrip_tile(const float* __restrict__ x, float* __restrict__ out) {
  const int tid  = threadIdx.x;
  const int lane = tid & 31;
  const int wvu  = __builtin_amdgcn_readfirstlane(tid >> 5);  // uniform wave id
  const int img  = blockIdx.y;
  const int tr   = blockIdx.x >> 3;   // 8x8 tiles of 64x64 over 512x512
  const int tc   = blockIdx.x & 7;

  __shared__ __align__(16) float    sIn[INR * INS];      // 80x80 input region
  __shared__ __align__(16) _Float16 sDh[3][32][HSTR];    // down bands, [q][p], zero-pad
  __shared__ __align__(16) _Float16 sTh[3][32][HSTR];    // pass-1 result, [row][col]
  __shared__ float sAA[32 * TS];                         // LL1 tile (256-res)
  __shared__ float sU[3][32 * TS];                       // upsampled detail tiles

  const float* __restrict__ xin = x + (size_t)img * IMGW * IMGW;

  // ---- stage 1: async-copy 80x80 input region to LDS (block-granular clamp)
  {
    const uint32_t inBase = (uint32_t)(uintptr_t)(void*)&sIn[0];
    for (int j = tid; j < INR * 20; j += 256) {
      int r  = j / 20, cg = j % 20;                 // row 0..79, 4-col group
      int rg = r >> 2, rr = r & 3;
      int sr = min(max(16 * tr + rg - 2, 0), 127) * 4 + rr;
      int sc = min(max(16 * tc + cg - 2, 0), 127) * 4;
      uint64_t gaddr = (uint64_t)(uintptr_t)(xin + sr * IMGW + sc);
      uint32_t laddr = inBase + (uint32_t)((r * INS + cg * 4) * 4);
      asm volatile("global_load_async_to_lds_b128 %0, %1, off"
                   :: "v"(laddr), "v"(gaddr) : "memory");
    }
    // overlap: zero the down-band pad regions while the DMA is in flight
    uint32_t* z = (uint32_t*)&sDh[0][0][0];
    for (int j = tid; j < (3 * 32 * HSTR) / 2; j += 256) z[j] = 0u;
    asm volatile("s_wait_asynccnt 0" ::: "memory");
  }
  __syncthreads();

  // ---- stage 2: level-1 Haar + 2x2 area mean -> transposed f16 down tiles
  for (int j = tid; j < DH * DH; j += 256) {
    int p = j / DH, q = j % DH;
    const float* bp = &sIn[(4 * p) * INS + 4 * q];
    float lhs = 0.f, hls = 0.f, hhs = 0.f;
#pragma unroll
    for (int r2 = 0; r2 < 2; ++r2)
#pragma unroll
      for (int c2 = 0; c2 < 2; ++c2) {
        float x00 = bp[(2 * r2) * INS + 2 * c2];
        float x01 = bp[(2 * r2) * INS + 2 * c2 + 1];
        float x10 = bp[(2 * r2 + 1) * INS + 2 * c2];
        float x11 = bp[(2 * r2 + 1) * INS + 2 * c2 + 1];
        float a0 = (x00 + x01) * SQ, d0 = (x00 - x01) * SQ;
        float a1 = (x10 + x11) * SQ, d1 = (x10 - x11) * SQ;
        float aa = (a0 + a1) * SQ,  lh = (a0 - a1) * SQ;
        float hl = (d0 + d1) * SQ,  hh = (d0 - d1) * SQ;
        lhs += lh; hls += hl; hhs += hh;
        if (p >= 2 && p < 18 && q >= 2 && q < 18)
          sAA[(2 * (p - 2) + r2) * TS + 2 * (q - 2) + c2] = aa;
      }
    sDh[0][q][p] = (_Float16)(lhs * 0.25f);   // transposed: [col q][row p]
    sDh[1][q][p] = (_Float16)(hls * 0.25f);
    sDh[2][q][p] = (_Float16)(hhs * 0.25f);
  }
  __syncthreads();

  // ---- stage 3 (WMMA pass 1): rows.  T(32x32) = Wr(32x32K) @ D(32Kx32)
  // 12 jobs = 3 bands x 2 row-chunks x 2 col-chunks, uniform wave split.
  for (int job = wvu; job < 12; job += 8) {
    int band = job >> 2, cr = (job >> 1) & 1, cc = job & 1;
    v16h A = *(const v16h*)cWrA.v[cr][lane];              // banded weights
    int nq = 16 * cc + (lane & 15);
    int Kb = (lane < 16) ? 0 : 16;
    v16h B = *(const v16h*)&sDh[band][nq][Kb];            // zero-padded data
    v8f  C = {};
    C = __builtin_amdgcn_wmma_f32_16x16x32_f16(false, A, false, B,
                                               (short)0, C, false, false);
    int row0 = 16 * cr + ((lane >= 16) ? 8 : 0);
    int col  = 16 * cc + (lane & 15);
#pragma unroll
    for (int v = 0; v < 8; ++v) sTh[band][row0 + v][col] = (_Float16)C[v];
  }
  __syncthreads();

  // ---- stage 4 (WMMA pass 2): cols.  U(32x32) = T(32x32K) @ Wc(32Kx32)
  for (int job = wvu; job < 12; job += 8) {
    int band = job >> 2, cr = (job >> 1) & 1, cc = job & 1;
    int i  = (lane & 15) + 16 * cr;
    int kb = (lane < 16) ? 0 : 8;
    v8h lo = *(const v8h*)&sTh[band][i][kb];              // K 0..7 / 8..15
    v8h hi = *(const v8h*)&sTh[band][i][kb + 16];         // K 16..23 / 24..31
    v16h A = __builtin_shufflevector(lo, hi, 0, 1, 2, 3, 4, 5, 6, 7,
                                             8, 9, 10, 11, 12, 13, 14, 15);
    v16h B = *(const v16h*)cWcB.v[cc][lane];              // banded weights
    v8f  C = {};
    C = __builtin_amdgcn_wmma_f32_16x16x32_f16(false, A, false, B,
                                               (short)0, C, false, false);
    int row0 = 16 * cr + ((lane >= 16) ? 8 : 0);
    int col  = 16 * cc + (lane & 15);
#pragma unroll
    for (int v = 0; v < 8; ++v) sU[band][(row0 + v) * TS + col] = C[v];
  }
  __syncthreads();

  // ---- stage 5: level-1 inverse Haar, coalesced float2 stores
  float* __restrict__ op = out + (size_t)img * IMGW * IMGW;
  const int oy = 64 * tr, ox = 64 * tc;
  for (int qi = tid; qi < 32 * 32; qi += 256) {
    int r = qi >> 5, c = qi & 31;
    float aa = sAA[r * TS + c];
    float lh = sU[0][r * TS + c];
    float hl = sU[1][r * TS + c];
    float hh = sU[2][r * TS + c];
    float a0 = (aa + lh) * SQ, a1 = (aa - lh) * SQ;
    float d0 = (hl + hh) * SQ, d1 = (hl - hh) * SQ;
    float2 top = make_float2((a0 + d0) * SQ, (a0 - d0) * SQ);
    float2 bot = make_float2((a1 + d1) * SQ, (a1 - d1) * SQ);
    *(float2*)&op[(size_t)(oy + 2 * r) * IMGW + ox + 2 * c]     = top;
    *(float2*)&op[(size_t)(oy + 2 * r + 1) * IMGW + ox + 2 * c] = bot;
  }
}

extern "C" void kernel_launch(void* const* d_in, const int* in_sizes, int n_in,
                              void* d_out, int out_size, void* d_ws, size_t ws_size,
                              hipStream_t stream) {
  (void)n_in; (void)out_size; (void)d_ws; (void)ws_size;
  const float* x = (const float*)d_in[0];
  float* out = (float*)d_out;
  int nimg = in_sizes[0] / (IMGW * IMGW);   // B*C = 96 for (32,3,512,512)
  dim3 grid(64, nimg);                      // 8x8 output tiles per image
  dwt_roundtrip_tile<<<grid, 256, 0, stream>>>(x, out);
}
